// HebbianConvBlock_10428180594726
// MI455X (gfx1250) — compile-verified, matrix operand order checked by reading, not run on hardware
//
#include <hip/hip_runtime.h>
#include <hip/hip_bf16.h>
#include <math.h>

// ---------------- CDNA5 WMMA plumbing (wave32) ----------------
typedef __attribute__((ext_vector_type(16))) __bf16 v16bf;
typedef __attribute__((ext_vector_type(8)))  float  v8f;
typedef __attribute__((ext_vector_type(4)))  unsigned uint32x4;
typedef __attribute__((ext_vector_type(8)))  int      int32x8;
typedef __attribute__((ext_vector_type(4)))  int      int32x4;

static __device__ inline v8f wmma_bf16(v16bf a, v16bf b, v8f c) {
  // D = A(16x32) * B(32x16) + C(16x16), f32 accumulate
  return __builtin_amdgcn_wmma_f32_16x16x32_bf16(
      /*neg_a=*/false, a, /*neg_b=*/false, b,
      /*c_mod=*/(short)0, c, /*reuse_a=*/false, /*reuse_b=*/false);
}

// Load a 16x32 bf16 A/B fragment for this lane from LDS.
// Per ISA 7.12.2 (16-bit A 16x32): lane holds row (lane&15); VGPRs 0-3 hold
// 8 consecutive K starting at (lane>=16 ? 8 : 0); VGPRs 4-7 hold 8 consecutive
// K starting 16 later.
static __device__ inline v16bf load_frag(const __bf16* p) {
  union { v16bf v; uint4 q[2]; } u;
  u.q[0] = *(const uint4*)(p);
  u.q[1] = *(const uint4*)(p + 16);
  return u.v;
}

#define LDP 40   // padded LDS row stride (bf16 elems): 80B, 16B-aligned chunks

// ---------------- Tensor Data Mover (gfx1250 TDM) ----------------
#if __has_builtin(__builtin_amdgcn_tensor_load_to_lds) && \
    __has_builtin(__builtin_amdgcn_s_wait_tensorcnt)
#define USE_TDM 1
#else
#define USE_TDM 0
#endif

#if USE_TDM
// Issue one TDM descriptor load of a 2D tile (tile_w x tile_h elements of
// 2-byte data, global row stride = stride_elems) into LDS at lds_addr, with
// LDS padding of 4 DWORDs (16B) after every 16 DWORDs (64B row) so the tile
// lands with an 80B (LDP=40 elem) row stride.
static __device__ inline void tdm_load_tile_2d_bf16(unsigned lds_addr,
                                                    const void* gptr,
                                                    unsigned tile_w, unsigned tile_h,
                                                    unsigned stride_elems) {
  const unsigned long long ga = (unsigned long long)gptr;
  uint32x4 g0;
  g0[0] = 1u;                                       // count=1, user descriptor
  g0[1] = lds_addr;                                 // LDS byte address
  g0[2] = (unsigned)(ga & 0xFFFFFFFFu);             // global_addr[31:0]
  g0[3] = (unsigned)((ga >> 32) & 0x1FFFFFFu)       // global_addr[56:32]
        | (2u << 30);                               // type = 2 ("image")
  int32x8 g1;
  // data_size=1 (2B), pad_enable, pad_interval=3 (16 DW), pad_amount=3 (4 DW)
  g1[0] = (int)((1u << 16) | (1u << 20) | (3u << 22) | (3u << 25));
  g1[1] = (int)((tile_w & 0xFFFFu) << 16);          // tensor_dim0 = tile_w (lo)
  g1[2] = (int)((tile_w >> 16) | ((tile_h & 0xFFFFu) << 16)); // dim0 hi, dim1 lo
  g1[3] = (int)((tile_h >> 16) | ((tile_w & 0xFFFFu) << 16)); // dim1 hi, tile_dim0
  g1[4] = (int)(tile_h & 0xFFFFu);                  // tile_dim1 (tile_dim2=0)
  g1[5] = (int)stride_elems;                        // tensor_dim0_stride[31:0]
  g1[6] = 0;                                        // stride hi / dim1 stride
  g1[7] = 0;
  const int32x4 z4 = {0, 0, 0, 0};
#if defined(__clang_major__) && (__clang_major__ >= 23)
  const int32x8 z8 = {0, 0, 0, 0, 0, 0, 0, 0};
  __builtin_amdgcn_tensor_load_to_lds(g0, g1, z4, z4, z8, 0);
#else
  __builtin_amdgcn_tensor_load_to_lds(g0, g1, z4, z4, 0);
#endif
}
#endif  // USE_TDM

// ---------------- Generic GEMM: C[M,N] = A[M,K] @ W[N,K]^T ----------------
// mode 0: Cf = result (f32)   mode 1: Cf += result (residual add)
// mode 2: Cb = result (bf16, feeds a later WMMA GEMM)
__global__ __launch_bounds__(256)
void gemm_bf16_kernel(const __bf16* __restrict__ A, const __bf16* __restrict__ W,
                      float* __restrict__ Cf, __bf16* __restrict__ Cb,
                      int M, int N, int K, int mode) {
  __shared__ __bf16 As[128 * LDP];
  __shared__ __bf16 Ws[128 * LDP];

  const int tid  = threadIdx.x;
  const int lane = tid & 31;
  const int wave = tid >> 5;          // 8 waves
  const int wm   = wave >> 2;         // 0..1 -> 64 rows each
  const int wn   = wave & 3;          // 0..3 -> 32 cols each
  const int hi   = lane >> 4;
  const int ln   = lane & 15;

  const int blockN0 = blockIdx.x * 128;
  const int blockM0 = blockIdx.y * 128;

  const v8f vzero = {0.f,0.f,0.f,0.f,0.f,0.f,0.f,0.f};
  v8f acc[4][2];
#pragma unroll
  for (int i = 0; i < 4; ++i)
#pragma unroll
    for (int j = 0; j < 2; ++j) acc[i][j] = vzero;

#if USE_TDM
  const unsigned ldsA = (unsigned)(unsigned long long)(const void*)As;
  const unsigned ldsW = (unsigned)(unsigned long long)(const void*)Ws;
#else
  const int ldr = tid >> 1;           // 0..127 : staged row
  const int ldo = (tid & 1) * 16;     // 0 / 16 : staged col offset
#endif

  for (int k0 = 0; k0 < K; k0 += 32) {
#if USE_TDM
    // Tensor Data Mover stages both 128x32 bf16 tiles; one issue per block.
    if (wave == 0) {
      tdm_load_tile_2d_bf16(ldsA, A + (size_t)blockM0 * K + k0, 32, 128, K);
      tdm_load_tile_2d_bf16(ldsW, W + (size_t)blockN0 * K + k0, 32, 128, K);
      __builtin_amdgcn_s_wait_tensorcnt(0);
    }
#else
    const __bf16* ga = A + (size_t)(blockM0 + ldr) * K + k0 + ldo;
    const __bf16* gw = W + (size_t)(blockN0 + ldr) * K + k0 + ldo;
    *(uint4*)&As[ldr * LDP + ldo]     = *(const uint4*)ga;
    *(uint4*)&As[ldr * LDP + ldo + 8] = *(const uint4*)(ga + 8);
    *(uint4*)&Ws[ldr * LDP + ldo]     = *(const uint4*)gw;
    *(uint4*)&Ws[ldr * LDP + ldo + 8] = *(const uint4*)(gw + 8);
    if (k0 + 32 < K) {                 // gfx1250 global_prefetch_b8 path
      __builtin_prefetch(ga + 32, 0, 1);
      __builtin_prefetch(gw + 32, 0, 1);
    }
#endif
    __syncthreads();

    v16bf afrag[4], bfrag[2];
#pragma unroll
    for (int t = 0; t < 4; ++t)
      afrag[t] = load_frag(&As[(wm * 64 + t * 16 + ln) * LDP + hi * 8]);
#pragma unroll
    for (int t = 0; t < 2; ++t)
      bfrag[t] = load_frag(&Ws[(wn * 32 + t * 16 + ln) * LDP + hi * 8]);
#pragma unroll
    for (int tm = 0; tm < 4; ++tm)
#pragma unroll
      for (int tn = 0; tn < 2; ++tn)
        acc[tm][tn] = wmma_bf16(afrag[tm], bfrag[tn], acc[tm][tn]);
    __syncthreads();
  }

  // C layout (ISA 7.12.2): lane(0-15): M=r, N=lane; lane(16-31): M=8+r, N=lane-16
#pragma unroll
  for (int tm = 0; tm < 4; ++tm)
#pragma unroll
    for (int tn = 0; tn < 2; ++tn)
#pragma unroll
      for (int r = 0; r < 8; ++r) {
        const int row = blockM0 + wm * 64 + tm * 16 + r + hi * 8;
        const int col = blockN0 + wn * 32 + tn * 16 + ln;
        const float v = acc[tm][tn][r];
        const size_t idx = (size_t)row * N + col;
        if (mode == 0)      Cf[idx] = v;
        else if (mode == 1) Cf[idx] += v;
        else                Cb[idx] = (__bf16)v;
      }
}

// ---------------- Fused decayed memory attention ----------------
// reads[b,j,m] = sum_{i<=j} exp(min(cum[j]-cum[i],0)) *
//               (vals_i . q_j)/sqrt(DMEM) * keys[i,m] / sqrt(T)
// One block = (batch b, 64 j's). Loop 64-i tiles flash-attention style.
__global__ __launch_bounds__(256)
void decay_attn_kernel(const __bf16* __restrict__ vals,
                       const __bf16* __restrict__ queries,
                       const __bf16* __restrict__ keys,
                       const float*  __restrict__ cum,
                       __bf16* __restrict__ reads, int T) {
  __shared__ __bf16 Vs[64 * LDP];      // vals chunk   (64i x 32m)
  __shared__ __bf16 Qs[64 * LDP];      // queries chunk(64j x 32m)
  __shared__ __bf16 St[64 * 72];       // decayed scores, TRANSPOSED [j][i]
  __shared__ __bf16 KsT[256 * 72];     // keys tile TRANSPOSED [m][i]

  const int tid  = threadIdx.x;
  const int lane = tid & 31;
  const int wave = tid >> 5;
  const int hi   = lane >> 4;
  const int ln   = lane & 15;
  // phase A map (S: 64i x 64j):
  const int wiA = wave >> 1;           // 0..3 -> one 16-row i tile
  const int wjA = wave & 1;            // 0..1 -> 32 j
  // phase B map (reads: 64j x 256m):
  const int wjB = wave >> 2;           // 0..1 -> 32 j
  const int wmB = wave & 3;            // 0..3 -> 64 m

  const int b  = blockIdx.y;
  const int j0 = blockIdx.x * 64;
  const size_t rowB = (size_t)b * T;
  const float scale = 1.0f / (16.0f * 45.254834f);  // 1/(sqrt(256)*sqrt(2048))

  const v8f vzero = {0.f,0.f,0.f,0.f,0.f,0.f,0.f,0.f};
  v8f acc2[2][4];
#pragma unroll
  for (int a = 0; a < 2; ++a)
#pragma unroll
    for (int c = 0; c < 4; ++c) acc2[a][c] = vzero;

  for (int i0 = 0; i0 <= j0; i0 += 64) {
    // stage keys[i0..i0+63][:] transposed -> KsT[m][i]  (coalesced reads)
    for (int t = 0; t < 64; ++t)
      KsT[tid * 72 + t] = keys[(rowB + i0 + t) * 256 + tid];

    // ---- phase A: S = vals_tile @ queries_tile^T over m=256 ----
    v8f s[2] = {vzero, vzero};
    const int r8 = tid >> 2;           // 0..63
    const int o8 = (tid & 3) * 8;      // 0,8,16,24
    for (int k0 = 0; k0 < 256; k0 += 32) {
      *(uint4*)&Vs[r8 * LDP + o8] =
          *(const uint4*)&vals[(rowB + i0 + r8) * 256 + k0 + o8];
      *(uint4*)&Qs[r8 * LDP + o8] =
          *(const uint4*)&queries[(rowB + j0 + r8) * 256 + k0 + o8];
      __syncthreads();
      v16bf a  = load_frag(&Vs[(wiA * 16 + ln) * LDP + hi * 8]);
      v16bf b0 = load_frag(&Qs[(wjA * 32 + ln) * LDP + hi * 8]);
      v16bf b1 = load_frag(&Qs[(wjA * 32 + 16 + ln) * LDP + hi * 8]);
      s[0] = wmma_bf16(a, b0, s[0]);
      s[1] = wmma_bf16(a, b1, s[1]);
      __syncthreads();
    }

    // apply causal decay, store transposed bf16 tile for phase B
#pragma unroll
    for (int tn = 0; tn < 2; ++tn)
#pragma unroll
      for (int r = 0; r < 8; ++r) {
        const int il = wiA * 16 + r + hi * 8;
        const int jl = wjA * 32 + tn * 16 + ln;
        const int ig = i0 + il, jg = j0 + jl;
        float v = s[tn][r] * scale;
        float d = (ig <= jg)
                    ? expf(fminf(cum[rowB + jg] - cum[rowB + ig], 0.f))
                    : 0.f;
        St[jl * 72 + il] = (__bf16)(v * d);
      }
    __syncthreads();

    // ---- phase B: reads_tile += S^T(j x i) @ keys^T-staged(i x m) ----
#pragma unroll
    for (int ks = 0; ks < 64; ks += 32) {
      v16bf a2[2], b2[4];
#pragma unroll
      for (int tj = 0; tj < 2; ++tj)
        a2[tj] = load_frag(&St[(wjB * 32 + tj * 16 + ln) * 72 + ks + hi * 8]);
#pragma unroll
      for (int tm = 0; tm < 4; ++tm)
        b2[tm] = load_frag(&KsT[(wmB * 64 + tm * 16 + ln) * 72 + ks + hi * 8]);
#pragma unroll
      for (int tj = 0; tj < 2; ++tj)
#pragma unroll
        for (int tm = 0; tm < 4; ++tm)
          acc2[tj][tm] = wmma_bf16(a2[tj], b2[tm], acc2[tj][tm]);
    }
    __syncthreads();
  }

#pragma unroll
  for (int tj = 0; tj < 2; ++tj)
#pragma unroll
    for (int tm = 0; tm < 4; ++tm)
#pragma unroll
      for (int r = 0; r < 8; ++r) {
        const int jg = j0 + wjB * 32 + tj * 16 + r + hi * 8;
        const int mg = wmB * 64 + tm * 16 + ln;
        reads[(rowB + jg) * 256 + mg] = (__bf16)acc2[tj][tm][r];
      }
}

// ---------------- small VALU kernels ----------------
__global__ __launch_bounds__(256)
void rmsnorm_bf16_kernel(const float* __restrict__ x, const float* __restrict__ w,
                         __bf16* __restrict__ out, int D) {
  __shared__ float red[256];
  const int row = blockIdx.x, tid = threadIdx.x;
  const float* xr = x + (size_t)row * D;
  float ss = 0.f;
  for (int i = tid; i < D; i += 256) { float v = xr[i]; ss += v * v; }
  red[tid] = ss; __syncthreads();
  for (int off = 128; off > 0; off >>= 1) {
    if (tid < off) red[tid] += red[tid + off];
    __syncthreads();
  }
  const float inv = rsqrtf(red[0] / (float)D + 1e-6f);
  __bf16* orow = out + (size_t)row * D;
  for (int i = tid; i < D; i += 256) orow[i] = (__bf16)(xr[i] * inv * w[i]);
}

__global__ __launch_bounds__(256)
void gate_logit_kernel(const __bf16* __restrict__ hm, const float* __restrict__ wg,
                       const float* __restrict__ bg, float* __restrict__ logg, int D) {
  __shared__ float red[256];
  const int row = blockIdx.x, tid = threadIdx.x;
  const __bf16* hr = hm + (size_t)row * D;
  float s = 0.f;
  for (int i = tid; i < D; i += 256) s += (float)hr[i] * wg[i];
  red[tid] = s; __syncthreads();
  for (int off = 128; off > 0; off >>= 1) {
    if (tid < off) red[tid] += red[tid + off];
    __syncthreads();
  }
  if (tid == 0) {
    float g = 1.f / (1.f + expf(-(red[0] + bg[0])));
    logg[row] = logf(g + 1e-8f);
  }
}

__global__ __launch_bounds__(256)
void cumsum_kernel(const float* __restrict__ lg, float* __restrict__ cum, int T) {
  __shared__ float part[256];
  const int b = blockIdx.x, tid = threadIdx.x;
  const int chunk = 8;                      // T=2048 / 256
  const float* src = lg + (size_t)b * T + tid * chunk;
  float loc[8], s = 0.f;
  for (int i = 0; i < chunk; ++i) { s += src[i]; loc[i] = s; }
  part[tid] = s; __syncthreads();
  for (int off = 1; off < 256; off <<= 1) {
    float v = (tid >= off) ? part[tid - off] : 0.f;
    __syncthreads();
    part[tid] += v;
    __syncthreads();
  }
  const float base = (tid > 0) ? part[tid - 1] : 0.f;
  float* dst = cum + (size_t)b * T + tid * chunk;
  for (int i = 0; i < chunk; ++i) dst[i] = base + loc[i];
}

__global__ __launch_bounds__(256)
void conv_gate_kernel(const float* __restrict__ gv, const float* __restrict__ wconv,
                      __bf16* __restrict__ out, int T, int D) {
  const size_t idx = (size_t)blockIdx.x * 256 + threadIdx.x;  // over B*T*D
  const int d = (int)(idx % D);
  const size_t bt = idx / D;
  const int t = (int)(bt % T);
  const float g = 1.f / (1.f + expf(-gv[bt * 2 * D + d]));
  float acc = 0.f;
#pragma unroll
  for (int k = 0; k < 15; ++k) {
    const int ts = t - 14 + k;
    if (ts >= 0)
      acc += gv[(bt - (size_t)(t - ts)) * 2 * D + D + d] * wconv[d * 15 + k];
  }
  out[idx] = (__bf16)(acc * g);
}

__global__ __launch_bounds__(256)
void swiglu_kernel(const __bf16* __restrict__ g, const __bf16* __restrict__ u,
                   __bf16* __restrict__ out, size_t n) {
  const size_t i = (size_t)blockIdx.x * 256 + threadIdx.x;
  if (i < n) {
    const float gf = (float)g[i];
    out[i] = (__bf16)(gf / (1.f + expf(-gf)) * (float)u[i]);
  }
}

__global__ __launch_bounds__(256)
void cvt_bf16_kernel(const float* __restrict__ in, __bf16* __restrict__ out, size_t n) {
  const size_t i = (size_t)blockIdx.x * 256 + threadIdx.x;
  if (i < n) out[i] = (__bf16)in[i];
}

__global__ __launch_bounds__(256)
void copy_f32_kernel(const float* __restrict__ in, float* __restrict__ out, size_t n) {
  const size_t i = (size_t)blockIdx.x * 256 + threadIdx.x;
  if (i < n) out[i] = in[i];
}

// ---------------- host orchestration ----------------
extern "C" void kernel_launch(void* const* d_in, const int* in_sizes, int n_in,
                              void* d_out, int out_size, void* d_ws, size_t ws_size,
                              hipStream_t stream) {
  constexpr int Bn = 4, T = 2048, D = 1024, DFF = 4096, DMEM = 256;
  constexpr int M = Bn * T;                 // 8192 rows

  const float* x       = (const float*)d_in[0];
  const float* w_ln1   = (const float*)d_in[1];
  const float* w_up    = (const float*)d_in[2];   // (2D, D)
  const float* w_conv  = (const float*)d_in[3];   // (D,1,15)
  const float* w_down  = (const float*)d_in[4];   // (D, D)
  const float* w_lnm   = (const float*)d_in[5];
  const float* w_key   = (const float*)d_in[6];   // (DMEM, D)
  const float* w_val   = (const float*)d_in[7];
  const float* w_qry   = (const float*)d_in[8];
  const float* w_gate  = (const float*)d_in[9];   // (1, D)
  const float* b_gate  = (const float*)d_in[10];  // (1,)
  const float* w_mout  = (const float*)d_in[11];  // (D, DMEM)
  const float* w_ln2   = (const float*)d_in[12];
  const float* Wg      = (const float*)d_in[13];  // (DFF, D)
  const float* Wu      = (const float*)d_in[14];
  const float* Wo      = (const float*)d_in[15];  // (D, DFF)
  float* xout = (float*)d_out;                    // residual stream lives here

  // workspace bump allocator
  char* ws = (char*)d_ws; size_t off = 0;
  auto alloc = [&](size_t bytes) -> void* {
    void* p = ws + off; off = (off + bytes + 255) & ~(size_t)255; return p;
  };
  __bf16* hbuf   = (__bf16*)alloc((size_t)M * D * 2);         // h / hm / h2
  float*  gv     = (float* )alloc((size_t)M * 2 * D * 4);
  __bf16* mixin  = (__bf16*)alloc((size_t)M * D * 2);
  __bf16* keysb  = (__bf16*)alloc((size_t)M * DMEM * 2);
  __bf16* valsb  = (__bf16*)alloc((size_t)M * DMEM * 2);
  __bf16* qrysb  = (__bf16*)alloc((size_t)M * DMEM * 2);
  float*  logg   = (float* )alloc((size_t)M * 4);
  float*  cum    = (float* )alloc((size_t)M * 4);
  __bf16* readsb = (__bf16*)alloc((size_t)M * DMEM * 2);
  __bf16* gbuf   = (__bf16*)alloc((size_t)M * DFF * 2);
  __bf16* ubuf   = (__bf16*)alloc((size_t)M * DFF * 2);
  __bf16* midbuf = (__bf16*)alloc((size_t)M * DFF * 2);
  __bf16* wupb   = (__bf16*)alloc((size_t)2 * D * D * 2);
  __bf16* wdnb   = (__bf16*)alloc((size_t)D * D * 2);
  __bf16* wkb    = (__bf16*)alloc((size_t)DMEM * D * 2);
  __bf16* wvb    = (__bf16*)alloc((size_t)DMEM * D * 2);
  __bf16* wqb    = (__bf16*)alloc((size_t)DMEM * D * 2);
  __bf16* wmob   = (__bf16*)alloc((size_t)D * DMEM * 2);
  __bf16* wgb    = (__bf16*)alloc((size_t)DFF * D * 2);
  __bf16* wub    = (__bf16*)alloc((size_t)DFF * D * 2);
  __bf16* wob    = (__bf16*)alloc((size_t)D * DFF * 2);

  auto cvt = [&](const float* src, __bf16* dst, size_t n) {
    cvt_bf16_kernel<<<dim3((unsigned)(n / 256)), 256, 0, stream>>>(src, dst, n);
  };
  auto gemm = [&](const __bf16* A, const __bf16* W, float* Cf, __bf16* Cb,
                  int m, int n, int k, int mode) {
    gemm_bf16_kernel<<<dim3(n / 128, m / 128), 256, 0, stream>>>(A, W, Cf, Cb, m, n, k, mode);
  };

  // weights -> bf16
  cvt(w_up,   wupb, (size_t)2 * D * D);
  cvt(w_down, wdnb, (size_t)D * D);
  cvt(w_key,  wkb,  (size_t)DMEM * D);
  cvt(w_val,  wvb,  (size_t)DMEM * D);
  cvt(w_qry,  wqb,  (size_t)DMEM * D);
  cvt(w_mout, wmob, (size_t)D * DMEM);
  cvt(Wg,     wgb,  (size_t)DFF * D);
  cvt(Wu,     wub,  (size_t)DFF * D);
  cvt(Wo,     wob,  (size_t)D * DFF);

  // residual stream = x
  copy_f32_kernel<<<dim3((unsigned)((size_t)M * D / 256)), 256, 0, stream>>>(
      x, xout, (size_t)M * D);

  // --- mixer ---
  rmsnorm_bf16_kernel<<<M, 256, 0, stream>>>(xout, w_ln1, hbuf, D);
  gemm(hbuf, wupb, gv, nullptr, M, 2 * D, D, 0);
  conv_gate_kernel<<<dim3((unsigned)((size_t)M * D / 256)), 256, 0, stream>>>(
      gv, w_conv, mixin, T, D);
  gemm(mixin, wdnb, xout, nullptr, M, D, D, 1);          // residual add

  // --- decayed memory ---
  rmsnorm_bf16_kernel<<<M, 256, 0, stream>>>(xout, w_lnm, hbuf, D);
  gemm(hbuf, wkb, nullptr, keysb, M, DMEM, D, 2);
  gemm(hbuf, wvb, nullptr, valsb, M, DMEM, D, 2);
  gemm(hbuf, wqb, nullptr, qrysb, M, DMEM, D, 2);
  gate_logit_kernel<<<M, 256, 0, stream>>>(hbuf, w_gate, b_gate, logg, D);
  cumsum_kernel<<<Bn, 256, 0, stream>>>(logg, cum, T);
  decay_attn_kernel<<<dim3(T / 64, Bn), 256, 0, stream>>>(
      valsb, qrysb, keysb, cum, readsb, T);
  gemm(readsb, wmob, xout, nullptr, M, D, DMEM, 1);      // residual add

  // --- SwiGLU MLP ---
  rmsnorm_bf16_kernel<<<M, 256, 0, stream>>>(xout, w_ln2, hbuf, D);
  gemm(hbuf, wgb, nullptr, gbuf, M, DFF, D, 2);
  gemm(hbuf, wub, nullptr, ubuf, M, DFF, D, 2);
  swiglu_kernel<<<dim3((unsigned)((size_t)M * DFF / 256)), 256, 0, stream>>>(
      gbuf, ubuf, midbuf, (size_t)M * DFF);
  gemm(midbuf, wob, xout, nullptr, M, D, DFF, 1);        // residual add
}
